// Gate_833223655781
// MI455X (gfx1250) — matrix operand order repaired and checked by reading, not compiled, not run on hardware
//
#include <hip/hip_runtime.h>

#define N_TOKENS 131072
#define HIDDEN   2048
#define EXPERTS  64
#define TOPK     8

#define KCHUNK   128            // K elements staged in LDS per outer iteration
#define BROW     (KCHUNK + 8)   // padded LDS row stride (bf16 elems): 272B -> rows offset 4 banks

typedef __attribute__((ext_vector_type(16))) __bf16 v16bf;
typedef __attribute__((ext_vector_type(8)))  __bf16 v8bf;
typedef __attribute__((ext_vector_type(8)))  float  v8f;
typedef __attribute__((ext_vector_type(4)))  float  f32x4;

typedef unsigned long long u64;
typedef unsigned int       u32;

__device__ __forceinline__ void ce_max_lo(u64& a, u64& b) {
    // compare-exchange: keep max at a, min at b (static indices only)
    const u64 mx = (a < b) ? b : a;
    const u64 mn = (a < b) ? a : b;
    a = mx; b = mn;
}

__global__ __launch_bounds__(256) void moe_gate_kernel(
    const float* __restrict__ x,      // [N_TOKENS, HIDDEN]
    const float* __restrict__ eemb,   // [EXPERTS, HIDDEN]
    const float* __restrict__ bias,   // [EXPERTS]
    float* __restrict__ out_w,        // [N_TOKENS, TOPK]
    int*   __restrict__ out_i,        // [N_TOKENS, TOPK]
    float* __restrict__ out_s)        // [N_TOKENS, EXPERTS]
{
    __shared__ alignas(16) __bf16 ldsB[EXPERTS * BROW];  // 17408 B: bf16 E chunk (later reused for top-k keys)
    __shared__ float ldsS[128 * 65];                     // 33280 B: scores for the block's 128 tokens

    const int tid  = threadIdx.x;
    const int wave = tid >> 5;
    const int lane = tid & 31;
    const int l16  = lane & 15;
    const int lhi  = lane >> 4;               // 0 or 1 (half-wave)
    const int tok0 = blockIdx.x * 128;        // first token of this block
    const int wtok = tok0 + wave * 16;        // first token of this wave's M-tile

    v8f c[4] = {};                            // 16 tokens x 64 experts accumulators

    // ---- register-buffered prefetch of E chunks (software pipeline) ----
    // E (512 KB) is re-read by every block: keep default (temporal) policy -> L2 resident.
    f32x4 pre[8];                             // 4 units x 8 floats
    auto load_chunk = [&](int ko) {
        #pragma unroll
        for (int i = 0; i < 4; ++i) {
            const int u  = tid + i * 256;     // 0..1023
            const int e  = u >> 4;            // expert row 0..63
            const int kk = (u & 15) * 8;      // 0..120
            const f32x4* gp = reinterpret_cast<const f32x4*>(
                eemb + (size_t)e * HIDDEN + ko + kk);
            pre[2 * i]     = gp[0];
            pre[2 * i + 1] = gp[1];
        }
    };
    load_chunk(0);

    for (int ko = 0; ko < HIDDEN; ko += KCHUNK) {
        // ---- stage: cvt prefetched regs -> bf16 -> LDS ----
        #pragma unroll
        for (int i = 0; i < 4; ++i) {
            const int u  = tid + i * 256;
            const int e  = u >> 4;
            const int kk = (u & 15) * 8;
            const f32x4 f0 = pre[2 * i];
            const f32x4 f1 = pre[2 * i + 1];
            v8bf bv;
            bv[0] = (__bf16)f0[0]; bv[1] = (__bf16)f0[1];
            bv[2] = (__bf16)f0[2]; bv[3] = (__bf16)f0[3];
            bv[4] = (__bf16)f1[0]; bv[5] = (__bf16)f1[1];
            bv[6] = (__bf16)f1[2]; bv[7] = (__bf16)f1[3];
            *reinterpret_cast<v8bf*>(&ldsB[e * BROW + kk]) = bv;
        }
        __syncthreads();

        // issue next chunk's global loads now; they complete under the WMMA work
        if (ko + KCHUNK < HIDDEN) load_chunk(ko + KCHUNK);

        // ---- compute: 4 K-steps of 32, 4 N-tiles each ----
        v16bf af[4];   // 4-deep A fragment rotation: rewrite distance = 16 WMMAs, no WAR nops
        #pragma unroll
        for (int ks = 0; ks < KCHUNK; ks += 32) {
            const int kg = ko + ks;
            // A fragment (ISA 16-bit A 16x32 layout):
            // lanes 0-15 hold K {0..7,16..23}; lanes 16-31 hold K {8..15,24..31}
            // X is streamed exactly once -> non-temporal loads (TH=NT).
            const float* arow = x + (size_t)(wtok + l16) * HIDDEN + kg + lhi * 8;
            const f32x4* ap = reinterpret_cast<const f32x4*>(arow);
            const f32x4 a0 = __builtin_nontemporal_load(ap);
            const f32x4 a1 = __builtin_nontemporal_load(ap + 1);
            const f32x4* ap2 = reinterpret_cast<const f32x4*>(arow + 16);
            const f32x4 a2 = __builtin_nontemporal_load(ap2);
            const f32x4 a3 = __builtin_nontemporal_load(ap2 + 1);
            v16bf& a = af[ks >> 5];
            a[0]  = (__bf16)a0[0]; a[1]  = (__bf16)a0[1]; a[2]  = (__bf16)a0[2]; a[3]  = (__bf16)a0[3];
            a[4]  = (__bf16)a1[0]; a[5]  = (__bf16)a1[1]; a[6]  = (__bf16)a1[2]; a[7]  = (__bf16)a1[3];
            a[8]  = (__bf16)a2[0]; a[9]  = (__bf16)a2[1]; a[10] = (__bf16)a2[2]; a[11] = (__bf16)a2[3];
            a[12] = (__bf16)a3[0]; a[13] = (__bf16)a3[1]; a[14] = (__bf16)a3[2]; a[15] = (__bf16)a3[3];

            #pragma unroll
            for (int t = 0; t < 4; ++t) {
                // B fragment: lane = column n = t*16 + l16, K {8h..8h+7, 16+8h..16+8h+7} bf16
                const __bf16* bp = &ldsB[(t * 16 + l16) * BROW + lhi * 8 + ks];
                const v8bf blo = *reinterpret_cast<const v8bf*>(bp);
                const v8bf bhi = *reinterpret_cast<const v8bf*>(bp + 16);
                v16bf b;
                #pragma unroll
                for (int r = 0; r < 8; ++r) { b[r] = blo[r]; b[8 + r] = bhi[r]; }
                c[t] = __builtin_amdgcn_wmma_f32_16x16x32_bf16(
                    false, a, false, b, (short)0, c[t], false, false);
            }
        }
        __syncthreads();   // protect LDS before next chunk overwrite
    }

    // ---- epilogue: sigmoid + bias, scatter scores to LDS + global ----
    float bv[4];
    #pragma unroll
    for (int t = 0; t < 4; ++t) bv[t] = bias[t * 16 + l16];

    #pragma unroll
    for (int t = 0; t < 4; ++t) {
        #pragma unroll
        for (int r = 0; r < 8; ++r) {
            // C layout: VGPR r, lanes 0-15: M=r; lanes 16-31: M=8+r; N = l16
            const float s   = 1.0f / (1.0f + __expf(-c[t][r])) + bv[t];
            const int  tokl = wave * 16 + lhi * 8 + r;   // token within block
            const int  col  = t * 16 + l16;              // expert
            ldsS[tokl * 65 + col] = s;
            __builtin_nontemporal_store(s, &out_s[(size_t)(tok0 + tokl) * EXPERTS + col]);
        }
    }
    __syncthreads();

    // ---- top-8, phase A: 2 threads/token, each selects sorted top-8 of 32 ----
    // keys: sign-folded float bits (monotonic u32) << 6 | (63 - expert), stored in reused ldsB
    u64* ldsK = reinterpret_cast<u64*>(ldsB);            // 256*8*8 = 16384 B <= 17408 B
    {
        const int tl   = tid >> 1;                       // token local 0..127
        const int half = tid & 1;                        // which 32 experts
        const float* row = &ldsS[tl * 65 + half * 32];
        u32 uv[32];
        #pragma unroll
        for (int j = 0; j < 32; ++j) {
            const u32 b = __float_as_uint(row[j]);
            uv[j] = (b & 0x80000000u) ? ~b : (b | 0x80000000u);
        }
        u32 chosen = 0;
        #pragma unroll
        for (int i = 0; i < TOPK; ++i) {
            u32 best = 0u;
            int bj = 0;
            #pragma unroll
            for (int j = 0; j < 32; ++j) {
                const bool free = ((chosen >> j) & 1u) == 0u;
                if (free && uv[j] > best) { best = uv[j]; bj = j; }
            }
            chosen |= (1u << bj);
            const int jg = half * 32 + bj;
            ldsK[tid * TOPK + i] = ((u64)best << 6) | (u64)(63 - jg);
        }
    }
    __syncthreads();

    // ---- top-8, phase B: bitonic merge of two sorted 8-lists (static indices) ----
    if (tid < 128) {
        u64 s[16];
        #pragma unroll
        for (int i = 0; i < 8; ++i) s[i]     = ldsK[(2 * tid)     * TOPK + i];       // descending
        #pragma unroll
        for (int i = 0; i < 8; ++i) s[8 + i] = ldsK[(2 * tid + 1) * TOPK + (7 - i)]; // ascending
        // s[0..15] is bitonic; split: low half = top-8 set
        #pragma unroll
        for (int i = 0; i < 8; ++i) ce_max_lo(s[i], s[i + 8]);
        // sort the (bitonic) low half descending: strides 4, 2, 1
        ce_max_lo(s[0], s[4]); ce_max_lo(s[1], s[5]); ce_max_lo(s[2], s[6]); ce_max_lo(s[3], s[7]);
        ce_max_lo(s[0], s[2]); ce_max_lo(s[1], s[3]); ce_max_lo(s[4], s[6]); ce_max_lo(s[5], s[7]);
        ce_max_lo(s[0], s[1]); ce_max_lo(s[2], s[3]); ce_max_lo(s[4], s[5]); ce_max_lo(s[6], s[7]);

        float tv[TOPK];
        int   ti[TOPK];
        float sum = 0.0f;
        #pragma unroll
        for (int i = 0; i < TOPK; ++i) {
            const u64 k   = s[i];
            const int idx = 63 - (int)(k & 63ull);
            const u32 u   = (u32)(k >> 6);
            const u32 bits = (u & 0x80000000u) ? (u & 0x7fffffffu) : ~u;
            const float f = __uint_as_float(bits);
            tv[i] = f; ti[i] = idx; sum += f;
        }
        const int   tok = tok0 + tid;
        const float inv = 1.0f / sum;
        #pragma unroll
        for (int i = 0; i < TOPK; ++i) {
            __builtin_nontemporal_store(tv[i] * inv, &out_w[(size_t)tok * TOPK + i]);
            __builtin_nontemporal_store(ti[i],       &out_i[(size_t)tok * TOPK + i]);
        }
    }
}

extern "C" void kernel_launch(void* const* d_in, const int* in_sizes, int n_in,
                              void* d_out, int out_size, void* d_ws, size_t ws_size,
                              hipStream_t stream) {
    const float* x = (const float*)d_in[0];
    const float* e = (const float*)d_in[1];
    const float* b = (const float*)d_in[2];

    float* out   = (float*)d_out;
    float* out_w = out;                                          // [N, 8] fp32
    int*   out_i = (int*)(out + (size_t)N_TOKENS * TOPK);        // [N, 8] int32 (bit-identical region)
    float* out_s = out + (size_t)N_TOKENS * TOPK * 2;            // [N, 64] fp32

    dim3 grid(N_TOKENS / 128);
    dim3 block(256);
    moe_gate_kernel<<<grid, block, 0, stream>>>(x, e, b, out_w, out_i, out_s);
}
